// CausalAttention_40845138985317
// MI455X (gfx1250) — matrix operand order repaired and checked by reading, not compiled, not run on hardware
//
#include <hip/hip_runtime.h>

// Chunkwise causal linear attention for MI455X (gfx1250, wave32, WMMA).
// All matmuls run on v_wmma_f32_16x16x32_bf16 (fp32 accumulate).
// Workspace layout (bf16 unless noted): xb, w{q,k,v,o}b, q, k, kT, vT, oacc(f32), ob
// -> ~130 MB, fits inside the 192 MB global L2, so the multi-pass pipeline
// stays on-die; HBM traffic is ~70 MB (x + out + weights).

#define BATCH  4
#define TLEN   4096
#define DIM    512
#define NCHUNK 32
#define CLEN   128            // TLEN / NCHUNK
#define MTOT   (BATCH * TLEN) // 16384

typedef __bf16 bf16;
typedef bf16  v16bf __attribute__((ext_vector_type(16)));
typedef bf16  v8bf  __attribute__((ext_vector_type(8)));
typedef bf16  v4bf  __attribute__((ext_vector_type(4)));
typedef float v8f   __attribute__((ext_vector_type(8)));
typedef float v4f   __attribute__((ext_vector_type(4)));
typedef int   v4i   __attribute__((ext_vector_type(4)));

#define AS1 __attribute__((address_space(1)))
#define AS3 __attribute__((address_space(3)))

union FragU { v16bf v; v8bf h[2]; };

// A-fragment (16x32, M x K): lane (r = lane&15, h = lane>>4) holds row r,
// K = h*8 .. h*8+7 and h*8+16 .. h*8+23  -> two contiguous 16B chunks.
__device__ __forceinline__ v16bf load_a_frag(const bf16* base, int ld) {
  const int lane = threadIdx.x & 31;
  const int r = lane & 15, h = lane >> 4;
  FragU u;
  const bf16* p = base + (size_t)r * ld + h * 8;
  u.h[0] = *(const v8bf*)(p);
  u.h[1] = *(const v8bf*)(p + 16);
  return u.v;
}

// B-fragment (32x16, K x N): lane (n = lane&15, h = lane>>4) holds column n,
// K = h*8 .. h*8+7 and h*8+16 .. h*8+23 -> two contiguous 16B chunks of the row-of-B^T storage.
__device__ __forceinline__ v16bf load_b_frag(const bf16* base, int ld) {
  const int lane = threadIdx.x & 31;
  const int n = lane & 15, h = lane >> 4;
  FragU u;
  const bf16* p = base + (size_t)n * ld + h * 8;
  u.h[0] = *(const v8bf*)(p);
  u.h[1] = *(const v8bf*)(p + 16);
  return u.v;
}

#define WMMA_BF16(a, b, c) \
  __builtin_amdgcn_wmma_f32_16x16x32_bf16(false, (a), false, (b), (short)0, (c), false, false)

// 16-byte global -> LDS copy. Prefer the CDNA5 async DMA path
// (GLOBAL_LOAD_ASYNC_TO_LDS_B128, ASYNCcnt-tracked, no VGPR round trip);
// fall back to a plain load/store pair if the builtin is unavailable.
// Probe-derived signature: (v4i AS1* gsrc, v4i AS3* ldst, int offset, int cpol).
#if __has_builtin(__builtin_amdgcn_global_load_async_to_lds_b128)
#define HAVE_ASYNC_LDS 1
#else
#define HAVE_ASYNC_LDS 0
#endif

__device__ __forceinline__ void copy16_g2l(const bf16* g, bf16* l) {
#if HAVE_ASYNC_LDS
  __builtin_amdgcn_global_load_async_to_lds_b128(
      (AS1 v4i*)g, (AS3 v4i*)l, 0, 0);
#else
  *(v8bf*)l = *(const v8bf*)g;
#endif
}

__device__ __forceinline__ void wait_g2l() {
#if HAVE_ASYNC_LDS
#if __has_builtin(__builtin_amdgcn_s_wait_asynccnt)
  __builtin_amdgcn_s_wait_asynccnt(0);
#else
  asm volatile("s_wait_asynccnt 0" ::: "memory");
#endif
#endif
}

// ---------------------------------------------------------------- converts
// Every convert source is read exactly once -> nontemporal loads keep the
// 192 MB L2 free for the bf16 working set.
__global__ __launch_bounds__(256) void f32_to_bf16_kernel(
    const float* __restrict__ src, bf16* __restrict__ dst, int n4) {
  int i = blockIdx.x * 256 + threadIdx.x;
  if (i < n4) {
    v4f f = __builtin_nontemporal_load((const v4f*)(src + (size_t)i * 4));
    v4bf o;
    o[0] = (bf16)f[0]; o[1] = (bf16)f[1]; o[2] = (bf16)f[2]; o[3] = (bf16)f[3];
    *(v4bf*)(dst + (size_t)i * 4) = o;
  }
}

// ---------------------------------------------------------------- GEMM
// Y[m,n] = sum_k A[m,k] * W[n,k]   (nn.Linear: y = x @ W^T)
// Block: 128x128 tile, 8 waves in a 4(M) x 2(N) grid, 32x64 per wave,
// K staged through LDS (async DMA) in steps of 32. Optional outputs:
//   WR: bf16 row-major [M,N];  WT: bf16 transposed [B][N][Tdim];  WF: f32 row-major.
template <bool WR, bool WT, bool WF>
__global__ __launch_bounds__(256) void gemm_xwT_kernel(
    const bf16* __restrict__ A, const bf16* __restrict__ W,
    bf16* __restrict__ outR, bf16* __restrict__ outT, float* __restrict__ outF,
    int N, int K, int Tdim) {
  __shared__ bf16 As[128 * 40];   // padded row stride 40 to spread LDS banks
  __shared__ bf16 Bs[128 * 40];
  const int m0 = blockIdx.y * 128;
  const int n0 = blockIdx.x * 128;
  const int wave = threadIdx.x >> 5;
  const int wm = wave & 3, wn = wave >> 2;
  const int row = threadIdx.x >> 1, seg = (threadIdx.x & 1) * 16;
  const bf16* gA = A + (size_t)(m0 + row) * K + seg;
  const bf16* gB = W + (size_t)(n0 + row) * K + seg;
  bf16* lA = &As[row * 40 + seg];
  bf16* lB = &Bs[row * 40 + seg];

  v8f acc[2][4] = {};
  for (int k0 = 0; k0 < K; k0 += 32) {
    copy16_g2l(gA + k0,     lA);
    copy16_g2l(gA + k0 + 8, lA + 8);
    copy16_g2l(gB + k0,     lB);
    copy16_g2l(gB + k0 + 8, lB + 8);
    if (k0 + 32 < K) {  // hint next K-slab into cache: global_prefetch_b8
      __builtin_prefetch(gA + k0 + 32, 0, 1);
      __builtin_prefetch(gB + k0 + 32, 0, 1);
    }
    wait_g2l();
    __syncthreads();
    v16bf a0 = load_a_frag(&As[(wm * 32 + 0)  * 40], 40);
    v16bf a1 = load_a_frag(&As[(wm * 32 + 16) * 40], 40);
#pragma unroll
    for (int jj = 0; jj < 4; ++jj) {
      v16bf b = load_b_frag(&Bs[(wn * 64 + jj * 16) * 40], 40);
      acc[0][jj] = WMMA_BF16(a0, b, acc[0][jj]);
      acc[1][jj] = WMMA_BF16(a1, b, acc[1][jj]);
    }
    __syncthreads();
  }

  const int lane = threadIdx.x & 31;
  const int r = lane & 15, h = lane >> 4;
#pragma unroll
  for (int i = 0; i < 2; ++i)
#pragma unroll
    for (int jj = 0; jj < 4; ++jj) {
      const int nn = n0 + wn * 64 + jj * 16 + r;
      const int mb = m0 + wm * 32 + i * 16 + h * 8;
#pragma unroll
      for (int j = 0; j < 8; ++j) {
        const int m = mb + j;
        const float val = acc[i][jj][j];
        if (WR) outR[(size_t)m * N + nn] = (bf16)val;
        if (WT) outT[((size_t)(m / Tdim) * N + nn) * Tdim + (m % Tdim)] = (bf16)val;
        if (WF) __builtin_nontemporal_store(val, &outF[(size_t)m * N + nn]);
      }
    }
}

// ---------------------------------------------------------------- intra-chunk
// Per (b,c): scores = q k^T (128x128, K=512), causal mask, bf16 round-trip
// through LDS, then out_intra = scores @ v via vT rows as B-fragments.
__global__ __launch_bounds__(256) void intra_kernel(
    const bf16* __restrict__ qb, const bf16* __restrict__ kb,
    const bf16* __restrict__ vT, float* __restrict__ oacc) {
  __shared__ bf16 sS[CLEN * 136];  // scores tile, padded stride
  const int c = blockIdx.x, b = blockIdx.y;
  const int t0 = c * CLEN;
  const bf16* qc = qb + ((size_t)b * TLEN + t0) * DIM;
  const bf16* kc = kb + ((size_t)b * TLEN + t0) * DIM;
  const int wave = threadIdx.x >> 5;
  const int wm = wave & 3, wn = wave >> 2;
  const int lane = threadIdx.x & 31;
  const int r = lane & 15, h = lane >> 4;

  {  // phase 1: scores
    v8f s[2][4] = {};
    for (int k0 = 0; k0 < DIM; k0 += 32) {
      v16bf a0 = load_a_frag(qc + (size_t)(wm * 32 + 0)  * DIM + k0, DIM);
      v16bf a1 = load_a_frag(qc + (size_t)(wm * 32 + 16) * DIM + k0, DIM);
#pragma unroll
      for (int jj = 0; jj < 4; ++jj) {
        v16bf bfr = load_b_frag(kc + (size_t)(wn * 64 + jj * 16) * DIM + k0, DIM);
        s[0][jj] = WMMA_BF16(a0, bfr, s[0][jj]);
        s[1][jj] = WMMA_BF16(a1, bfr, s[1][jj]);
      }
    }
#pragma unroll
    for (int i = 0; i < 2; ++i)
#pragma unroll
      for (int jj = 0; jj < 4; ++jj)
#pragma unroll
        for (int j = 0; j < 8; ++j) {
          const int m = wm * 32 + i * 16 + h * 8 + j;  // t within chunk
          const int n = wn * 64 + jj * 16 + r;         // s within chunk
          const float val = (n <= m) ? s[i][jj][j] : 0.0f;  // causal mask
          sS[m * 136 + n] = (bf16)val;
        }
  }
  __syncthreads();

  // phase 2: out_intra = scores @ v  (K = 128 token dim; B-frags from vT rows)
  for (int nb = 0; nb < 4; ++nb) {
    v8f o[2][4] = {};
    for (int k0 = 0; k0 < CLEN; k0 += 32) {
      v16bf a0 = load_a_frag(&sS[(wm * 32 + 0)  * 136 + k0], 136);
      v16bf a1 = load_a_frag(&sS[(wm * 32 + 16) * 136 + k0], 136);
#pragma unroll
      for (int jj = 0; jj < 4; ++jj) {
        const bf16* vrow =
            vT + ((size_t)b * DIM + nb * 128 + wn * 64 + jj * 16) * TLEN + t0 + k0;
        v16bf bfr = load_b_frag(vrow, TLEN);
        o[0][jj] = WMMA_BF16(a0, bfr, o[0][jj]);
        o[1][jj] = WMMA_BF16(a1, bfr, o[1][jj]);
      }
    }
#pragma unroll
    for (int i = 0; i < 2; ++i)
#pragma unroll
      for (int jj = 0; jj < 4; ++jj)
#pragma unroll
        for (int j = 0; j < 8; ++j) {
          const int m = wm * 32 + i * 16 + h * 8 + j;
          const int n = nb * 128 + wn * 64 + jj * 16 + r;
          oacc[((size_t)b * TLEN + t0 + m) * DIM + n] = o[i][jj][j];
        }
  }
}

// ---------------------------------------------------------------- inter-chunk
// Per (b, 32-wide D slice): keep S^T slice [32 x 512] resident in LDS across
// the sequential chunk scan. Per chunk: out += q_c @ S (exclusive prefix),
// then S^T += v^T k  (both WMMA GEMMs; A/B fragments contiguous via vT/kT).
__global__ __launch_bounds__(256) void inter_kernel(
    const bf16* __restrict__ qb, const bf16* __restrict__ kT,
    const bf16* __restrict__ vT, float* __restrict__ oacc) {
  __shared__ bf16 ST[32 * 520];  // S^T[n][d], padded stride
  const int ns = blockIdx.x, b = blockIdx.y;
  const int n0 = ns * 32;
  const int wave = threadIdx.x >> 5;
  const int lane = threadIdx.x & 31;
  const int r = lane & 15, h = lane >> 4;
  const int wm  = wave & 3, wn  = wave >> 2;  // 4x2 over (128 t, 32 n)
  const int wm2 = wave & 1, wn2 = wave >> 1;  // 2x4 over (32 n, 512 d)

  for (int idx = threadIdx.x; idx < 32 * 520; idx += 256) ST[idx] = (bf16)0.0f;
  __syncthreads();

  for (int c = 0; c < NCHUNK; ++c) {
    const int t0 = c * CLEN;
    if (c > 0) {  // exclusive prefix: chunk 0 sees S == 0
      v8f acc[2] = {};
      for (int k0 = 0; k0 < DIM; k0 += 32) {
        v16bf a0 = load_a_frag(
            qb + ((size_t)b * TLEN + t0 + wm * 32 + 0)  * DIM + k0, DIM);
        v16bf a1 = load_a_frag(
            qb + ((size_t)b * TLEN + t0 + wm * 32 + 16) * DIM + k0, DIM);
        v16bf bfr = load_b_frag(&ST[(wn * 16) * 520 + k0], 520);
        acc[0] = WMMA_BF16(a0, bfr, acc[0]);
        acc[1] = WMMA_BF16(a1, bfr, acc[1]);
      }
#pragma unroll
      for (int i = 0; i < 2; ++i)
#pragma unroll
        for (int j = 0; j < 8; ++j) {
          const int m = wm * 32 + i * 16 + h * 8 + j;
          const int n = wn * 16 + r;
          const size_t off = ((size_t)b * TLEN + t0 + m) * DIM + n0 + n;
          oacc[off] += acc[i][j];
        }
    }
    __syncthreads();  // all S reads done before update writes

    v8f upd[8] = {};
    for (int k0 = 0; k0 < CLEN; k0 += 32) {
      v16bf a = load_a_frag(
          vT + ((size_t)b * DIM + n0 + wm2 * 16) * TLEN + t0 + k0, TLEN);
#pragma unroll
      for (int jj = 0; jj < 8; ++jj) {
        v16bf bfr = load_b_frag(
            kT + ((size_t)b * DIM + wn2 * 128 + jj * 16) * TLEN + t0 + k0, TLEN);
        upd[jj] = WMMA_BF16(a, bfr, upd[jj]);
      }
    }
#pragma unroll
    for (int jj = 0; jj < 8; ++jj)
#pragma unroll
      for (int j = 0; j < 8; ++j) {
        const int n = wm2 * 16 + h * 8 + j;
        const int d = wn2 * 128 + jj * 16 + r;
        const float cur = (float)ST[n * 520 + d];
        ST[n * 520 + d] = (bf16)(cur + upd[jj][j]);
      }
    __syncthreads();
  }
}

// ---------------------------------------------------------------- launch
extern "C" void kernel_launch(void* const* d_in, const int* in_sizes, int n_in,
                              void* d_out, int out_size, void* d_ws, size_t ws_size,
                              hipStream_t stream) {
  (void)in_sizes; (void)n_in; (void)out_size; (void)ws_size;
  const float* x  = (const float*)d_in[0];
  const float* wq = (const float*)d_in[1];
  const float* wk = (const float*)d_in[2];
  const float* wv = (const float*)d_in[3];
  const float* wo = (const float*)d_in[4];

  char* ws = (char*)d_ws;
  size_t off = 0;
  auto take = [&](size_t bytes) -> char* {
    char* p = ws + off;
    off += (bytes + 255) & ~(size_t)255;
    return p;
  };
  const size_t BTD = (size_t)MTOT * DIM;  // 8,388,608 elements
  const size_t DD  = (size_t)DIM * DIM;
  bf16*  xb   = (bf16*)take(BTD * 2);
  bf16*  wqb  = (bf16*)take(DD * 2);
  bf16*  wkb  = (bf16*)take(DD * 2);
  bf16*  wvb  = (bf16*)take(DD * 2);
  bf16*  wob  = (bf16*)take(DD * 2);
  bf16*  qbuf = (bf16*)take(BTD * 2);
  bf16*  kbuf = (bf16*)take(BTD * 2);
  bf16*  kTb  = (bf16*)take(BTD * 2);
  bf16*  vTb  = (bf16*)take(BTD * 2);
  float* oacc = (float*)take(BTD * 4);
  bf16*  obuf = (bf16*)take(BTD * 2);

  const int nx4 = (int)(BTD / 4);
  const int nw4 = (int)(DD / 4);
  f32_to_bf16_kernel<<<(nx4 + 255) / 256, 256, 0, stream>>>(x, xb, nx4);
  f32_to_bf16_kernel<<<(nw4 + 255) / 256, 256, 0, stream>>>(wq, wqb, nw4);
  f32_to_bf16_kernel<<<(nw4 + 255) / 256, 256, 0, stream>>>(wk, wkb, nw4);
  f32_to_bf16_kernel<<<(nw4 + 255) / 256, 256, 0, stream>>>(wv, wvb, nw4);
  f32_to_bf16_kernel<<<(nw4 + 255) / 256, 256, 0, stream>>>(wo, wob, nw4);

  dim3 gg(DIM / 128, MTOT / 128);
  // q: row-major only.  k: row-major + transposed.  v: transposed only.
  gemm_xwT_kernel<true, false, false><<<gg, 256, 0, stream>>>(
      xb, wqb, qbuf, nullptr, nullptr, DIM, DIM, TLEN);
  gemm_xwT_kernel<true, true, false><<<gg, 256, 0, stream>>>(
      xb, wkb, kbuf, kTb, nullptr, DIM, DIM, TLEN);
  gemm_xwT_kernel<false, true, false><<<gg, 256, 0, stream>>>(
      xb, wvb, nullptr, vTb, nullptr, DIM, DIM, TLEN);

  intra_kernel<<<dim3(NCHUNK, BATCH), 256, 0, stream>>>(qbuf, kbuf, vTb, oacc);
  inter_kernel<<<dim3(DIM / 32, BATCH), 256, 0, stream>>>(qbuf, kTb, vTb, oacc);

  f32_to_bf16_kernel<<<(nx4 + 255) / 256, 256, 0, stream>>>(oacc, obuf, nx4);
  gemm_xwT_kernel<false, false, true><<<gg, 256, 0, stream>>>(
      obuf, wob, nullptr, nullptr, (float*)d_out, DIM, DIM, TLEN);
}
